// SineGen_26912265076753
// MI455X (gfx1250) — matrix operand order repaired and checked by reading, not compiled, not run on hardware
//
#include <hip/hip_runtime.h>
#include <cstdint>
#include <cstddef>

typedef _Float16 f16;
typedef __attribute__((ext_vector_type(16))) _Float16 v16h;
typedef __attribute__((ext_vector_type(8)))  float    v8f;

#define BATCH 16
#define TLEN 65536
#define DH 9
#define CHUNK 2048
#define NCHUNK (TLEN / CHUNK)        // 32
#define THREADS 256
#define PERTHREAD (CHUNK / THREADS)  // 8
#define INV_SR (1.0f / 16000.0f)
#define SINE_AMP 0.1f
#define NOISE_STD 0.003f
#define UNVOICED_AMP (0.1f / 3.0f)

// mod-1 addition for values in [0,1): associative scan operator
__device__ __forceinline__ float fracadd(float a, float b) {
    float s = a + b;
    return (s >= 1.0f) ? (s - 1.0f) : s;
}
__device__ __forceinline__ float frac1(float x) { return x - floorf(x); }

// tanh(x) = 1 - 2/(exp(2x)+1); native v_exp + v_rcp, correct saturation at +/-inf
__device__ __forceinline__ float fast_tanh(float x) {
    float e = __expf(2.0f * x);
    return 1.0f - 2.0f * __builtin_amdgcn_rcpf(e + 1.0f);
}

// ---------------- Kernel 1: per-chunk mod-1 sums of f0/SR ----------------
__global__ __launch_bounds__(THREADS)
void k_chunksum(const float* __restrict__ f0, float* __restrict__ tot) {
    const int b = blockIdx.x / NCHUNK, c = blockIdx.x % NCHUNK;
    const int tid = threadIdx.x;
    const float* p = f0 + (size_t)b * TLEN + (size_t)c * CHUNK + (size_t)tid * PERTHREAD;
    float s = 0.0f;
    for (int j = 0; j < PERTHREAD; ++j) s = fracadd(s, p[j] * INV_SR);
    __shared__ float red[THREADS];
    red[tid] = s;
    __syncthreads();
    for (int off = THREADS / 2; off > 0; off >>= 1) {
        if (tid < off) red[tid] = fracadd(red[tid], red[tid + off]);
        __syncthreads();
    }
    if (tid == 0) tot[blockIdx.x] = red[0];
}

// ---------------- Kernel 2: per-batch exclusive scan of chunk sums ----------------
__global__ void k_chunkscan(const float* __restrict__ tot, float* __restrict__ pref) {
    const int tid = threadIdx.x;
    if (tid < BATCH) {
        float acc = 0.0f;
        for (int c = 0; c < NCHUNK; ++c) {
            float t = tot[tid * NCHUNK + c];
            pref[tid * NCHUNK + c] = acc;
            acc = fracadd(acc, t);
        }
    }
}

// ---------------- Kernel 3: main ----------------
struct Shared {
    float    f0[CHUNK];        // 8 KB, async-staged
    float    waveTot[8];
    float    outBuf[8][16];    // per-wave WMMA output bounce (512 B)
    unsigned tile[CHUNK * 5];  // 40 KB: per row 10 halves (9 sine_waves + pad)
};

__global__ __launch_bounds__(THREADS)
void k_main(const float* __restrict__ f0,
            const float* __restrict__ rinit,
            const float* __restrict__ noise,
            const float* __restrict__ W,
            const float* __restrict__ bias,
            const float* __restrict__ pref,
            float* __restrict__ out) {
    __shared__ Shared sh;
    const int b = blockIdx.x / NCHUNK, c = blockIdx.x % NCHUNK;
    const int tid = threadIdx.x;
    const int lane = tid & 31, wid = tid >> 5;

    // ---- async-stage f0 chunk (8 KB) into LDS: 2 x b128 per lane ----
    {
        const char* gsrc = (const char*)(f0 + (size_t)b * TLEN + (size_t)c * CHUNK);
        unsigned o0 = (unsigned)(size_t)(&sh.f0[0]) + (unsigned)tid * 16u;
        unsigned long long g0 = (unsigned long long)(size_t)gsrc + (unsigned long long)tid * 16ull;
        asm volatile("global_load_async_to_lds_b128 %0, %1, off" :: "v"(o0), "v"(g0) : "memory");
        unsigned o1 = o0 + 4096u;
        unsigned long long g1 = g0 + 4096ull;
        asm volatile("global_load_async_to_lds_b128 %0, %1, off" :: "v"(o1), "v"(g1) : "memory");
        asm volatile("s_wait_asynccnt 0x0" ::: "memory");
    }
    __syncthreads();

    // ---- thread-local inclusive mod-1 scan over 8 contiguous elements ----
    float r[PERTHREAD], s[PERTHREAD];
    {
        float acc = 0.0f;
        for (int j = 0; j < PERTHREAD; ++j) {
            r[j] = sh.f0[tid * PERTHREAD + j];
            acc = fracadd(acc, r[j] * INV_SR);
            s[j] = acc;
        }
    }

    // ---- wave32 inclusive scan of thread totals (shuffles) ----
    float incl = s[PERTHREAD - 1];
    for (int d = 1; d < 32; d <<= 1) {
        float n = __shfl_up(incl, d, 32);
        if (lane >= d) incl = fracadd(incl, n);
    }
    float exclLane = __shfl_up(incl, 1, 32);
    if (lane == 0) exclLane = 0.0f;
    if (lane == 31) sh.waveTot[wid] = incl;
    __syncthreads();
    if (tid == 0) {  // exclusive scan of the 8 wave totals
        float acc = 0.0f;
        for (int w = 0; w < 8; ++w) {
            float t = sh.waveTot[w];
            sh.waveTot[w] = acc;
            acc = fracadd(acc, t);
        }
    }
    __syncthreads();
    const float base = fracadd(fracadd(pref[b * NCHUNK + c], sh.waveTot[wid]), exclLane);

    float rk[DH];
    for (int k = 0; k < DH; ++k) rk[k] = rinit[b * DH + k];

    // ---- per-element: harmonic phases -> sine_waves -> f16 LDS tile ----
    for (int j = 0; j < PERTHREAD; ++j) {
        const int row = tid * PERTHREAD + j;
        const float S = fracadd(base, s[j]);
        const float uv = (r[j] > 0.0f) ? 1.0f : 0.0f;
        const float namp = (r[j] > 0.0f) ? NOISE_STD : UNVOICED_AMP;
        const size_t nb = ((size_t)b * TLEN + (size_t)c * CHUNK + (size_t)row) * DH;
        union { f16 h[10]; unsigned u[5]; } pk;
        for (int k = 0; k < DH; ++k) {
            float phase = frac1((float)(k + 1) * S + rk[k]);
            // v_sin_f32 computes sin(2*pi*x) natively
            float sine = __builtin_amdgcn_sinf(phase) * SINE_AMP;
            float sw = sine * uv + namp * noise[nb + k];
            pk.h[k] = (f16)sw;
        }
        pk.h[9] = (f16)0.0f;
        for (int q = 0; q < 5; ++q) sh.tile[row * 5 + q] = pk.u[q];
    }
    __syncthreads();

    // ---- WMMA projection: D = A(16x32 f16 sine_waves, K padded) x B(W repl.) + bias ----
    union Frag { v16h v; f16 h[16]; unsigned u[8]; };
    Frag bf;
    for (int q = 0; q < 8; ++q) bf.u[q] = 0u;
    if (lane < 16) {  // B layout: lanes 0-15 hold K=0..7 of column N=lane
        for (int k = 0; k < 8; ++k) bf.h[k] = (f16)W[k];
    } else {  // lanes 16-31 hold K=8..15 in elements 0..7 -> only K8 nonzero
        bf.h[0] = (f16)W[8];
    }
    const float bb = bias[0];
    const bool lo = (lane < 16);

    float* outp = out + (size_t)b * TLEN + (size_t)c * CHUNK;
    const int rowBase0 = wid * (CHUNK / 8);  // 256 rows per wave -> 16 tiles
    for (int tI = 0; tI < 16; ++tI) {
        const int rowBase = rowBase0 + tI * 16;
        // branchless A fragment: every lane loads its row's 5 dwords, selects
        const int row = rowBase + (lane & 15);
        unsigned d0 = sh.tile[row * 5 + 0];
        unsigned d1 = sh.tile[row * 5 + 1];
        unsigned d2 = sh.tile[row * 5 + 2];
        unsigned d3 = sh.tile[row * 5 + 3];
        unsigned d4 = sh.tile[row * 5 + 4];
        Frag a;
        a.u[0] = lo ? d0 : d4;  // lanes>=16: VGPR0 = K8,K9(pad)
        a.u[1] = lo ? d1 : 0u;
        a.u[2] = lo ? d2 : 0u;
        a.u[3] = lo ? d3 : 0u;
        a.u[4] = 0u; a.u[5] = 0u; a.u[6] = 0u; a.u[7] = 0u;
        v8f cacc;
        for (int q = 0; q < 8; ++q) cacc[q] = bb;  // bias folded into C
        cacc = __builtin_amdgcn_wmma_f32_16x16x32_f16(
            false, a.v, false, bf.v, (short)0, cacc, false, false);
        // D[m][n] identical for all n; lane0 holds M=0..7, lane16 holds M=8..15.
        // Bounce through per-wave LDS so 16 lanes do tanh+store in parallel.
        if (lane == 0 || lane == 16) {
            const int mbase = lane >> 1;  // 0 or 8
            for (int m = 0; m < 8; ++m) sh.outBuf[wid][mbase + m] = cacc[m];
        }
        asm volatile("s_wait_dscnt 0x0" ::: "memory");
        float v = sh.outBuf[wid][lane & 15];
        float t = fast_tanh(v);
        if (lo) outp[rowBase + (lane & 15)] = t;
    }
}

extern "C" void kernel_launch(void* const* d_in, const int* in_sizes, int n_in,
                              void* d_out, int out_size, void* d_ws, size_t ws_size,
                              hipStream_t stream) {
    const float* f0    = (const float*)d_in[0];
    const float* rinit = (const float*)d_in[1];
    const float* noise = (const float*)d_in[2];
    const float* W     = (const float*)d_in[3];
    const float* bias  = (const float*)d_in[4];
    float* out = (float*)d_out;
    float* tot  = (float*)d_ws;              // 512 floats
    float* pref = tot + BATCH * NCHUNK;      // 512 floats

    k_chunksum<<<dim3(BATCH * NCHUNK), dim3(THREADS), 0, stream>>>(f0, tot);
    k_chunkscan<<<dim3(1), dim3(64), 0, stream>>>(tot, pref);
    k_main<<<dim3(BATCH * NCHUNK), dim3(THREADS), 0, stream>>>(f0, rinit, noise, W, bias, pref, out);
}
